// CapsNet2D_8976481649047
// MI455X (gfx1250) — matrix operand order repaired and checked by reading, not compiled, not run on hardware
//
#include <hip/hip_runtime.h>
#include <hip/hip_bf16.h>
#include <cstdint>

// ---------------------------------------------------------------------------
// CDNA5 (gfx1250) CapsNet2D.
// Conv / deconv layers: implicit GEMM on v_wmma_f32_16x16x32_f16.
//  - weights pre-packed per layer into WMMA A-fragment order (f16)
//  - A fragments staged into LDS per 32-K chunk:
//      * K=5 convs: Tensor Data Mover (tensor_load_to_lds, TENSORcnt)
//      * K=4/K=1:   double-buffered global_load_async_to_lds_b128 (ASYNCcnt)
//    read back with ds_load_b128; shared by all 8 waves of the block
//  - each wave owns MT M-tiles (MT wmma per chunk) reusing one B fragment
// Routing / squash / softmax / concat / norm: VALU kernels (reference math).
// ---------------------------------------------------------------------------

typedef __attribute__((ext_vector_type(16))) _Float16     v16h;
typedef __attribute__((ext_vector_type(8)))  float        v8f;
typedef __attribute__((ext_vector_type(4)))  unsigned int v4u;
typedef __attribute__((ext_vector_type(8)))  int          v8i;
typedef __attribute__((ext_vector_type(4)))  int          v4i;

static inline int ceil_div_i(int a, int b) { return (a + b - 1) / b; }

#define WAVES_PER_BLOCK 8

#if defined(__HIP_DEVICE_COMPILE__) && __has_builtin(__builtin_amdgcn_tensor_load_to_lds)
#define HAVE_TDM 1
#else
#define HAVE_TDM 0
#endif

// ---------------------------------------------------------------------------
// 1-D TDM copy: nbytes (multiple of 8) from global to LDS offset lds_off.
// D# bit layout per CDNA5 ISA ch.8 (group0: count|lds_addr|global_addr|type=2;
// group1: data_size=3 (8B), tensor_dim0 = tile_dim0 = nbytes/8, 1-D tile).
// This toolchain's builtin is the 6-arg (clang-23 / therock) form:
//   (uint32x4 g0, int32x8 g1, int32x4 g2, int32x4 g3, int32x8 g4, i32 cpol)
// ---------------------------------------------------------------------------
#if HAVE_TDM
__device__ inline void tdm_load_1d(uint32_t lds_off, const void* gsrc,
                                   uint32_t nbytes)
{
    uint64_t ga = (uint64_t)(uintptr_t)gsrc;
    v4u g0;
    g0.x = 1u;                                   // count=1 (valid user D#)
    g0.y = lds_off;                              // lds_addr [63:32]
    g0.z = (uint32_t)ga;                         // global_addr [95:64]
    g0.w = (uint32_t)(ga >> 32) | (2u << 30);    // global_addr[56:32] | type=2
    uint32_t n8 = nbytes >> 3;                   // elements of 8 bytes
    v8i g1;
    g1[0] = (int)(3u << 16);                     // wg_mask=0, data_size=3 (8B)
    g1[1] = (int)((n8 & 0xFFFFu) << 16);         // tensor_dim0[15:0]
    g1[2] = (int)((n8 >> 16) & 0xFFFFu);         // tensor_dim0[31:16]; dim1=0
    g1[3] = (int)((n8 & 0xFFFFu) << 16);         // tile_dim0
    g1[4] = 0;                                   // tile_dim1=0, tile_dim2=0
    g1[5] = (int)n8;                             // tensor_dim0_stride (low)
    g1[6] = 0;
    g1[7] = 0;
    v4i z4 = {0, 0, 0, 0};
    v8i z8 = {0, 0, 0, 0, 0, 0, 0, 0};
    __builtin_amdgcn_tensor_load_to_lds(g0, g1, z4, z4, z8, 0);
}
#endif

// ---------------------------------------------------------------------------
// Weight packing: W (fp32 conv layout) -> Wp f16 fragments.
// Wp layout: [kt][mt][lane][16] halfs  (fragment = 1024 B, ISA 16x32 A layout)
// ---------------------------------------------------------------------------
template<int K, int TRANSPOSED>
__global__ void pack_weights_kernel(const float* __restrict__ W,
                                    _Float16* __restrict__ Wp,
                                    int Cin, int Cout, int MtTot, int KtCnt)
{
    int idx = blockIdx.x * blockDim.x + threadIdx.x;
    int total = MtTot * KtCnt * 512;
    if (idx >= total) return;
    int j    = idx & 15;
    int lane = (idx >> 4) & 31;
    int frag = idx >> 9;
    int mt   = frag % MtTot;
    int kt   = frag / MtTot;
    int m    = mt * 16 + (lane & 15);
    int half = lane >> 4;
    int v = j >> 1, e = j & 1;
    int kc = ((v >> 2) << 4) + half * 8 + ((v & 3) << 1) + e;
    int k  = kt * 32 + kc;
    float wv = 0.0f;
    int Ktot = Cin * K * K;
    if (m < Cout && k < Ktot) {
        int ci = k / (K * K);
        int r  = k - ci * (K * K);
        int kh = r / K;
        int kw = r - kh * K;
        size_t widx = TRANSPOSED
            ? ((((size_t)ci * Cout + m) * K + (K - 1 - kh)) * K + (K - 1 - kw))
            : ((((size_t)m * Cin + ci) * K + kh) * K + kw);
        wv = W[widx];
    }
    Wp[idx] = (_Float16)wv;
}

// ---------------------------------------------------------------------------
// Implicit GEMM conv / conv-transpose with WMMA.
// Normal:      ih = oh*stride - pad + kh
// Transposed:  dh = oh - (K-1) + kh; ih = dh/stride, valid iff dh>=0, dh%s==0
// ---------------------------------------------------------------------------
template<int K, int TRANSPOSED, int MT, int USE_TDM>
__global__ void __launch_bounds__(256)
conv_wmma_kernel(const float* __restrict__ X, const _Float16* __restrict__ Wp,
                 float* __restrict__ Y,
                 int Cin, int Cout, int stride, int pad,
                 int Hi, int Wi, int Ho, int Wo, int MtTot, int KtCnt)
{
    __shared__ __align__(32) unsigned char smemA[8192];   // 2 x MT x 1 KB

    const int g      = blockIdx.z;
    const int mtile0 = blockIdx.y * MT;
    const int m0     = mtile0 * 16;
    const int lane   = threadIdx.x & 31;            // wave32
    const int wave   = threadIdx.x >> 5;
    const int HW     = Ho * Wo;
    const int n0     = (blockIdx.x * WAVES_PER_BLOCK + wave) * 16;

    const int half = lane >> 4;
    const int l16  = lane & 15;
    const int Ktot = Cin * K * K;
    const float* Xg = X + (size_t)g * Cin * Hi * Wi;

    const int  n      = n0 + l16;
    const bool nvalid = (n < HW);
    const int  nc     = nvalid ? n : (HW - 1);
    const int  oh     = nc / Wo;
    const int  ow     = nc - oh * Wo;

    // L2 prefetch hint for this lane's neighborhood of the input image
    {
        size_t pf = (size_t)nc;
        if (pf >= (size_t)Hi * Wi) pf = 0;
        __builtin_prefetch(&Xg[pf], 0, 0);
    }

    const uint32_t smem_base = (uint32_t)(uintptr_t)
        (__attribute__((address_space(3))) unsigned char*)smemA;

    // per-wave slice of one chunk's A fragments (2*MT half-fragment slices)
    auto stage_async = [&](int kt, int buf) {
        if (wave < 2 * MT) {
            const int mt_local = wave >> 1;
            const int part     = wave & 1;
            const unsigned char* src = (const unsigned char*)Wp
                + ((size_t)kt * MtTot + (mtile0 + mt_local)) * 1024
                + part * 512 + lane * 16;
            uint32_t dst = smem_base + buf * 4096
                         + mt_local * 1024 + part * 512 + lane * 16;
            asm volatile("global_load_async_to_lds_b128 %0, %1, off"
                         :: "v"(dst), "v"(src) : "memory");
        }
    };

    v8f zero = {};
    v8f acc[MT];
    #pragma unroll
    for (int t = 0; t < MT; ++t) acc[t] = zero;

#if HAVE_TDM
    constexpr bool use_tdm = (USE_TDM != 0);
#else
    constexpr bool use_tdm = false;
#endif

    if (!use_tdm) stage_async(0, 0);

    for (int kt = 0; kt < KtCnt; ++kt) {
        int buf;
        if (use_tdm) {
#if HAVE_TDM
            // one TDM descriptor moves the whole MT x 1KB chunk strip
            if (wave == 0) {
                const unsigned char* src = (const unsigned char*)Wp
                    + ((size_t)kt * MtTot + mtile0) * 1024;
                tdm_load_1d(smem_base, src, (uint32_t)(MT * 1024));
                __builtin_amdgcn_s_wait_tensorcnt(0);
            }
#endif
            __syncthreads();
            buf = 0;
        } else {
            // double-buffered async pipeline: issue kt+1, wait for kt
            if (kt + 1 < KtCnt) {
                stage_async(kt + 1, (kt + 1) & 1);
                asm volatile("s_wait_asynccnt 0x1" ::: "memory");
            } else {
                asm volatile("s_wait_asynccnt 0x0" ::: "memory");
            }
            __syncthreads();
            buf = kt & 1;
        }

        // ---- B fragment gather: 32x16 (KxN), f32 -> f16 ----
        const int kk0 = kt * 32;
        v16h b;
        #pragma unroll
        for (int v = 0; v < 8; ++v) {
            #pragma unroll
            for (int e = 0; e < 2; ++e) {
                const int k = kk0 + ((v >> 2) << 4) + half * 8 + ((v & 3) << 1) + e;
                float xv = 0.0f;
                if (k < Ktot && nvalid) {
                    const int ci = k / (K * K);
                    const int r  = k - ci * (K * K);
                    const int kh = r / K;
                    const int kw = r - kh * K;
                    if (TRANSPOSED) {
                        const int dh = oh - (K - 1) + kh;
                        const int dw = ow - (K - 1) + kw;
                        if (dh >= 0 && dw >= 0) {
                            const int ih = dh / stride, rh = dh - ih * stride;
                            const int iw = dw / stride, rw = dw - iw * stride;
                            if (rh == 0 && rw == 0 && ih < Hi && iw < Wi)
                                xv = Xg[((size_t)ci * Hi + ih) * Wi + iw];
                        }
                    } else {
                        const int ih = oh * stride - pad + kh;
                        const int iw = ow * stride - pad + kw;
                        if (ih >= 0 && ih < Hi && iw >= 0 && iw < Wi)
                            xv = Xg[((size_t)ci * Hi + ih) * Wi + iw];
                    }
                }
                b[2 * v + e] = (_Float16)xv;
            }
        }

        // ---- MT WMMAs sharing one B fragment; A from LDS (ds_load_b128) ----
        #pragma unroll
        for (int t = 0; t < MT; ++t) {
            v16h a = *(const v16h*)(smemA + buf * 4096 + t * 1024 + lane * 32);
            acc[t] = __builtin_amdgcn_wmma_f32_16x16x32_f16(
                         false, a, false, b, (short)0, acc[t], false, false);
        }
        __syncthreads();   // protect LDS before next chunk overwrites it
    }

    // ---- store D tiles (VGPR r: M = base + r + half*8, N = n) ----
    float* Yg = Y + (size_t)g * Cout * HW;
    #pragma unroll
    for (int t = 0; t < MT; ++t) {
        #pragma unroll
        for (int r = 0; r < 8; ++r) {
            const int M = m0 + t * 16 + r + half * 8;
            if (M < Cout && nvalid)
                Yg[(size_t)M * HW + n] = acc[t][r];
        }
    }
}

// ---------------------------------------------------------------------------
// conv1: Cin=1, 5x5, pad 2, stride 1, + bias + relu  (tiny layer, VALU)
// ---------------------------------------------------------------------------
__global__ void conv1_relu_kernel(const float* __restrict__ X,
                                  const float* __restrict__ W,
                                  const float* __restrict__ Bb,
                                  float* __restrict__ Y,
                                  int B, int Co, int H, int Wd)
{
    int idx = blockIdx.x * blockDim.x + threadIdx.x;
    int total = B * Co * H * Wd;
    if (idx >= total) return;
    int w = idx % Wd; int t = idx / Wd;
    int h = t % H;    t /= H;
    int co = t % Co;  int b = t / Co;
    const float* Xb = X + (size_t)b * H * Wd;
    float s = Bb[co];
    #pragma unroll
    for (int kh = 0; kh < 5; ++kh) {
        int ih = h - 2 + kh;
        if (ih < 0 || ih >= H) continue;
        #pragma unroll
        for (int kw = 0; kw < 5; ++kw) {
            int iw = w - 2 + kw;
            if (iw < 0 || iw >= Wd) continue;
            s += Xb[(size_t)ih * Wd + iw] * W[(co * 5 + kh) * 5 + kw];
        }
    }
    Y[idx] = s > 0.0f ? s : 0.0f;
}

// ---------------------------------------------------------------------------
// Routing kernels
// ---------------------------------------------------------------------------
__global__ void fill_zero_kernel(float* __restrict__ p, int n)
{
    int i = blockIdx.x * blockDim.x + threadIdx.x;
    if (i < n) p[i] = 0.0f;
}

// cij = softmax over Co of bij;  bij/cij layout (B*Ci, Co, HW)
__global__ void softmax_co_kernel(const float* __restrict__ bij,
                                  float* __restrict__ cij,
                                  int BCi, int Co, int HW)
{
    int idx = blockIdx.x * blockDim.x + threadIdx.x;
    int total = BCi * HW;
    if (idx >= total) return;
    int hw = idx % HW; int bci = idx / HW;
    const float* bp = bij + (size_t)bci * Co * HW + hw;
    float m = -1e30f;
    for (int c = 0; c < Co; ++c) m = fmaxf(m, bp[(size_t)c * HW]);
    float e[8];
    float s = 0.0f;
    for (int c = 0; c < Co; ++c) { e[c] = expf(bp[(size_t)c * HW] - m); s += e[c]; }
    float inv = 1.0f / s;
    float* cp = cij + (size_t)bci * Co * HW + hw;
    for (int c = 0; c < Co; ++c) cp[(size_t)c * HW] = e[c] * inv;
}

// sj[b,co,po,hw] = sum_ci cij[b,ci,co,hw]*votes[b,ci,co,po,hw] + bias[co,po]
__global__ void sj_kernel(const float* __restrict__ votes,
                          const float* __restrict__ cij,
                          const float* __restrict__ bias,
                          float* __restrict__ sj,
                          int B, int Ci, int Co, int Po, int HW)
{
    int idx = blockIdx.x * blockDim.x + threadIdx.x;
    int total = B * Co * Po * HW;
    if (idx >= total) return;
    int hw = idx % HW; int t = idx / HW;
    int po = t % Po;   t /= Po;
    int co = t % Co;   int b = t / Co;
    float s = bias[co * Po + po];
    for (int ci = 0; ci < Ci; ++ci) {
        float c = cij[(((size_t)b * Ci + ci) * Co + co) * HW + hw];
        float v = votes[((((size_t)b * Ci + ci) * Co + co) * Po + po) * (size_t)HW + hw];
        s += c * v;
    }
    sj[idx] = s;
}

// in-place squash over Po:  v = n2*s / ((1+n2)*sqrt(n2))
__global__ void squash_kernel(float* __restrict__ sj, int BCo, int Po, int HW)
{
    int idx = blockIdx.x * blockDim.x + threadIdx.x;
    int total = BCo * HW;
    if (idx >= total) return;
    int hw = idx % HW; int bco = idx / HW;
    float* p = sj + (size_t)bco * Po * HW + hw;
    float n2 = 0.0f;
    for (int po = 0; po < Po; ++po) {
        float v = p[(size_t)po * HW];
        n2 += v * v;
    }
    float scale = n2 / ((1.0f + n2) * sqrtf(n2));
    for (int po = 0; po < Po; ++po)
        p[(size_t)po * HW] *= scale;
}

// bij[b,ci,co,hw] += sum_po votes[b,ci,co,po,hw] * vj[b,co,po,hw]
__global__ void bupdate_kernel(float* __restrict__ bij,
                               const float* __restrict__ votes,
                               const float* __restrict__ vj,
                               int B, int Ci, int Co, int Po, int HW)
{
    int idx = blockIdx.x * blockDim.x + threadIdx.x;
    int total = B * Ci * Co * HW;
    if (idx >= total) return;
    int hw = idx % HW; int t = idx / HW;
    int co = t % Co;   t /= Co;
    int ci = t % Ci;   int b = t / Ci;
    const float* vp = votes + ((((size_t)b * Ci + ci) * Co + co) * Po) * (size_t)HW + hw;
    const float* jj = vj + (((size_t)b * Co + co) * Po) * (size_t)HW + hw;
    float s = 0.0f;
    for (int po = 0; po < Po; ++po)
        s += vp[(size_t)po * HW] * jj[(size_t)po * HW];
    bij[idx] += s;
}

// concat skip (B,C1,Po,H,W) with center-cropped xin (B,C2,Po,Hx,Wx)
__global__ void concat_kernel(const float* __restrict__ skip,
                              const float* __restrict__ xin,
                              float* __restrict__ out,
                              int B, int C1, int C2, int Po, int H, int W,
                              int Hx, int Wx)
{
    int idx = blockIdx.x * blockDim.x + threadIdx.x;
    int Ct = C1 + C2;
    int total = B * Ct * Po * H * W;
    if (idx >= total) return;
    int w = idx % W; int t = idx / W;
    int h = t % H;   t /= H;
    int po = t % Po; t /= Po;
    int c = t % Ct;  int b = t / Ct;
    float v;
    if (c < C1) {
        v = skip[((((size_t)b * C1 + c) * Po + po) * H + h) * W + w];
    } else {
        int offH = (Hx - H) / 2, offW = (Wx - W) / 2;
        v = xin[((((size_t)b * C2 + (c - C1)) * Po + po) * Hx + (h + offH)) * Wx
                + (w + offW)];
    }
    out[idx] = v;
}

// out[b,hw] = || f[b,0,:,hw] ||_2 over Po
__global__ void norm_kernel(const float* __restrict__ f, float* __restrict__ out,
                            int B, int Po, int HW)
{
    int idx = blockIdx.x * blockDim.x + threadIdx.x;
    int total = B * HW;
    if (idx >= total) return;
    int hw = idx % HW; int b = idx / HW;
    const float* p = f + (size_t)b * Po * HW + hw;
    float s = 0.0f;
    for (int po = 0; po < Po; ++po) {
        float v = p[(size_t)po * HW];
        s += v * v;
    }
    out[idx] = sqrtf(s);
}

// ---------------------------------------------------------------------------
// Host-side helpers
// ---------------------------------------------------------------------------
template<int K, int T, int MT, int TDM>
static void launch_conv(const float* X, const float* W, _Float16* WP, float* Y,
                        int G, int Cin, int Cout, int stride, int pad,
                        int Hi, int Wi, int Ho, int Wo, hipStream_t s)
{
    int Ktot  = Cin * K * K;
    int KtCnt = ceil_div_i(Ktot, 32);
    int MtTot = Cout / 16;
    int ptotal = MtTot * KtCnt * 512;
    pack_weights_kernel<K, T><<<ceil_div_i(ptotal, 256), 256, 0, s>>>(
        W, WP, Cin, Cout, MtTot, KtCnt);
    dim3 grid(ceil_div_i(Ho * Wo, 16 * WAVES_PER_BLOCK), MtTot / MT, G);
    conv_wmma_kernel<K, T, MT, TDM><<<grid, 256, 0, s>>>(
        X, WP, Y, Cin, Cout, stride, pad, Hi, Wi, Ho, Wo, MtTot, KtCnt);
}

static void run_routing(const float* votes, const float* bias, float* out,
                        float* bij, float* cij,
                        int B, int Ci, int Co, int Po, int HW, int routings,
                        hipStream_t s)
{
    int nb = B * Ci * Co * HW;
    fill_zero_kernel<<<ceil_div_i(nb, 256), 256, 0, s>>>(bij, nb);
    for (int t = 0; t < routings; ++t) {
        int nsm = B * Ci * HW;
        softmax_co_kernel<<<ceil_div_i(nsm, 256), 256, 0, s>>>(bij, cij, B * Ci, Co, HW);
        int nsj = B * Co * Po * HW;
        sj_kernel<<<ceil_div_i(nsj, 256), 256, 0, s>>>(votes, cij, bias, out,
                                                       B, Ci, Co, Po, HW);
        int nsq = B * Co * HW;
        squash_kernel<<<ceil_div_i(nsq, 256), 256, 0, s>>>(out, B * Co, Po, HW);
        if (t + 1 < routings) {
            bupdate_kernel<<<ceil_div_i(nb, 256), 256, 0, s>>>(bij, votes, out,
                                                               B, Ci, Co, Po, HW);
        }
    }
}

// ---------------------------------------------------------------------------
// Entry point
// ---------------------------------------------------------------------------
extern "C" void kernel_launch(void* const* d_in, const int* in_sizes, int n_in,
                              void* d_out, int out_size, void* d_ws, size_t ws_size,
                              hipStream_t stream)
{
    (void)in_sizes; (void)n_in; (void)out_size; (void)ws_size;

    const float* x   = (const float*)d_in[0];
    const float* w1  = (const float*)d_in[1];
    const float* b1  = (const float*)d_in[2];
    const float* w2  = (const float*)d_in[3];   const float* cb2  = (const float*)d_in[4];
    const float* w3  = (const float*)d_in[5];   const float* cb3  = (const float*)d_in[6];
    const float* w4  = (const float*)d_in[7];   const float* cb4  = (const float*)d_in[8];
    const float* w5  = (const float*)d_in[9];   const float* cb5  = (const float*)d_in[10];
    const float* w6  = (const float*)d_in[11];  const float* cb6  = (const float*)d_in[12];
    const float* w7  = (const float*)d_in[13];  const float* cb7  = (const float*)d_in[14];
    const float* w8  = (const float*)d_in[15];  const float* cb8  = (const float*)d_in[16];
    const float* w9  = (const float*)d_in[17];  const float* cb9  = (const float*)d_in[18];
    const float* w10 = (const float*)d_in[19];  const float* cb10 = (const float*)d_in[20];
    const float* w11 = (const float*)d_in[21];  const float* cb11 = (const float*)d_in[22];
    const float* w12 = (const float*)d_in[23];  const float* cb12 = (const float*)d_in[24];
    const float* w13 = (const float*)d_in[25];  const float* cb13 = (const float*)d_in[26];
    float* out = (float*)d_out;

    // ---- workspace bump allocator (floats, 256B aligned) ----
    float* base = (float*)d_ws;
    size_t off = 0;
    auto alloc = [&](size_t n) -> float* {
        float* p = base + off;
        off += (n + 63) & ~(size_t)63;
        return p;
    };
    float*    V   = alloc(33554432);   // votes scratch (max: c9, 134 MB)
    float*    A0  = alloc(6291456);    // activation ping (max: d12 concat)
    float*    A1  = alloc(2097152);    // activation pong
    float*    R   = alloc(4260096);    // routed-deconv scratch (max: d12 routed)
    float*    C1b = alloc(2097152);    // skip c1  (2,1,16,256,256)
    float*    C3b = alloc(2097152);    // skip c3  (2,4,16,128,128)
    float*    C5b = alloc(2097152);    // skip c5  (2,8,32,64,64)
    float*    BIJ = alloc(1090048);    // routing logits
    float*    CIJ = alloc(1090048);    // routing coefficients
    _Float16* WP  = (_Float16*)alloc(262144);  // packed weights (524288 halfs)

    // ---- conv1 + relu : (2,1,256,256) -> (2,16,256,256) ----
    {
        int total = 2 * 16 * 256 * 256;
        conv1_relu_kernel<<<ceil_div_i(total, 256), 256, 0, stream>>>(
            x, w1, b1, C1b, 2, 16, 256, 256);
    }

    // ---- p2 : caps_conv, Ci=1 Pi=16 -> Co=2 Po=16, s2 p2, r=1 ----
    launch_conv<5, 0, 2, 1>(C1b, w2, WP, V, 2, 16, 32, 2, 2, 256, 256, 128, 128, stream);
    run_routing(V, cb2, A0, BIJ, CIJ, 2, 1, 2, 16, 128 * 128, 1, stream);

    // ---- c3 : Ci=2 Pi=16 -> Co=4 Po=16, s1 p2, r=3 ----
    launch_conv<5, 0, 4, 1>(A0, w3, WP, V, 4, 16, 64, 1, 2, 128, 128, 128, 128, stream);
    run_routing(V, cb3, C3b, BIJ, CIJ, 2, 2, 4, 16, 128 * 128, 3, stream);

    // ---- c4 : Ci=4 Pi=16 -> Co=4 Po=32, s2 p2 ----
    launch_conv<5, 0, 4, 1>(C3b, w4, WP, V, 8, 16, 128, 2, 2, 128, 128, 64, 64, stream);
    run_routing(V, cb4, A0, BIJ, CIJ, 2, 4, 4, 32, 64 * 64, 3, stream);

    // ---- c5 : Ci=4 Pi=32 -> Co=8 Po=32, s1 p2 ----
    launch_conv<5, 0, 4, 1>(A0, w5, WP, V, 8, 32, 256, 1, 2, 64, 64, 64, 64, stream);
    run_routing(V, cb5, C5b, BIJ, CIJ, 2, 4, 8, 32, 64 * 64, 3, stream);

    // ---- c6 : Ci=8 Pi=32 -> Co=8 Po=64, s2 p2 ----
    launch_conv<5, 0, 4, 1>(C5b, w6, WP, V, 16, 32, 512, 2, 2, 64, 64, 32, 32, stream);
    run_routing(V, cb6, A0, BIJ, CIJ, 2, 8, 8, 64, 32 * 32, 3, stream);

    // ---- c7 : Ci=8 Pi=64 -> Co=8 Po=32, s1 p2 ----
    launch_conv<5, 0, 4, 1>(A0, w7, WP, V, 16, 64, 256, 1, 2, 32, 32, 32, 32, stream);
    run_routing(V, cb7, A1, BIJ, CIJ, 2, 8, 8, 32, 32 * 32, 3, stream);

    // ---- d8 : caps_deconv, Ci=8 Pi=32 -> Co=8 Po=32, K=4 s2; 32->66 ----
    launch_conv<4, 1, 4, 0>(A1, w8, WP, V, 16, 32, 256, 2, 0, 32, 32, 66, 66, stream);
    run_routing(V, cb8, R, BIJ, CIJ, 2, 8, 8, 32, 66 * 66, 3, stream);
    {
        int total = 2 * 16 * 32 * 64 * 64;
        concat_kernel<<<ceil_div_i(total, 256), 256, 0, stream>>>(
            C5b, R, A0, 2, 8, 8, 32, 64, 64, 66, 66);
    }

    // ---- c9 : Ci=16 Pi=32 -> Co=8 Po=32, s1 p2 ----
    launch_conv<5, 0, 4, 1>(A0, w9, WP, V, 32, 32, 256, 1, 2, 64, 64, 64, 64, stream);
    run_routing(V, cb9, A1, BIJ, CIJ, 2, 16, 8, 32, 64 * 64, 3, stream);

    // ---- d10 : deconv, Ci=8 Pi=32 -> Co=4 Po=16, K=4 s2; 64->130 ----
    launch_conv<4, 1, 4, 0>(A1, w10, WP, V, 16, 32, 64, 2, 0, 64, 64, 130, 130, stream);
    run_routing(V, cb10, R, BIJ, CIJ, 2, 8, 4, 16, 130 * 130, 3, stream);
    {
        int total = 2 * 8 * 16 * 128 * 128;
        concat_kernel<<<ceil_div_i(total, 256), 256, 0, stream>>>(
            C3b, R, A0, 2, 4, 4, 16, 128, 128, 130, 130);
    }

    // ---- c11 : Ci=8 Pi=16 -> Co=4 Po=16, s1 p2 ----
    launch_conv<5, 0, 4, 1>(A0, w11, WP, V, 16, 16, 64, 1, 2, 128, 128, 128, 128, stream);
    run_routing(V, cb11, A1, BIJ, CIJ, 2, 8, 4, 16, 128 * 128, 3, stream);

    // ---- d12 : deconv, Ci=4 Pi=16 -> Co=2 Po=16, K=4 s2; 128->258 ----
    launch_conv<4, 1, 2, 0>(A1, w12, WP, V, 8, 16, 32, 2, 0, 128, 128, 258, 258, stream);
    run_routing(V, cb12, R, BIJ, CIJ, 2, 4, 2, 16, 258 * 258, 3, stream);
    {
        int total = 2 * 3 * 16 * 256 * 256;
        concat_kernel<<<ceil_div_i(total, 256), 256, 0, stream>>>(
            C1b, R, A0, 2, 1, 2, 16, 256, 256, 258, 258);
    }

    // ---- f13 : Ci=3 Pi=16 -> Co=1 Po=16, 1x1, p0, r=3 ----
    launch_conv<1, 0, 1, 0>(A0, w13, WP, V, 6, 16, 16, 1, 0, 256, 256, 256, 256, stream);
    run_routing(V, cb13, A1, BIJ, CIJ, 2, 3, 1, 16, 256 * 256, 3, stream);

    // ---- final norm over Po -> (2,1,256,256) ----
    {
        int total = 2 * 256 * 256;
        norm_kernel<<<ceil_div_i(total, 256), 256, 0, stream>>>(
            A1, out, 2, 16, 256 * 256);
    }
}
